// GATNetwork_4707284157160
// MI455X (gfx1250) — matrix operand order repaired and checked by reading, not compiled, not run on hardware
//
#include <hip/hip_runtime.h>
#include <math.h>

typedef __bf16 bf16_t;
typedef __attribute__((ext_vector_type(16))) __bf16 v16bf;
typedef __attribute__((ext_vector_type(8)))  float  v8f;

#define NEG_SLOPE 0.2f
#define EPS_BN 1e-5f

// ---------------------------------------------------------------------------
// Utility kernels
// ---------------------------------------------------------------------------
__global__ void fill_kernel(float* __restrict__ p, size_t n, float v) {
    size_t i = (size_t)blockIdx.x * blockDim.x + threadIdx.x;
    if (i < n) p[i] = v;
}

__global__ void reduce_sum_kernel(const float* __restrict__ x, int n, float* __restrict__ out) {
    __shared__ float sh[256];
    float s = 0.f;
    for (size_t i = (size_t)blockIdx.x * blockDim.x + threadIdx.x; i < (size_t)n;
         i += (size_t)gridDim.x * blockDim.x)
        s += x[i];
    sh[threadIdx.x] = s;
    __syncthreads();
    for (int o = 128; o > 0; o >>= 1) {
        if ((int)threadIdx.x < o) sh[threadIdx.x] += sh[threadIdx.x + o];
        __syncthreads();
    }
    if (threadIdx.x == 0) atomicAdd(out, sh[0]);
}

__device__ __forceinline__ void atomicMaxFloat(float* addr, float val) {
    if (val >= 0.f)
        atomicMax((int*)addr, __float_as_int(val));
    else
        atomicMin((unsigned int*)addr, __float_as_uint(val));
}

// fp32 -> bf16 (RNE via v_cvt), streaming
__global__ void cvt_bf16_kernel(const float* __restrict__ x, bf16_t* __restrict__ y, size_t n) {
    size_t i = (size_t)blockIdx.x * blockDim.x + threadIdx.x;
    if (i < n) y[i] = (bf16_t)x[i];
}

// fp32 W[K x M] -> bf16 Wt[M x K] (transposed so B fragments are contiguous)
__global__ void cvt_bf16_transpose_kernel(const float* __restrict__ W, bf16_t* __restrict__ Wt,
                                          int K, int M) {
    size_t idx = (size_t)blockIdx.x * blockDim.x + threadIdx.x;
    if (idx >= (size_t)K * M) return;
    int k = (int)(idx / M);
    int m = (int)(idx % M);
    Wt[(size_t)m * K + k] = (bf16_t)W[idx];
}

// ---------------------------------------------------------------------------
// WMMA GEMM:  Y[n x M] = Xb[n x K](bf16) @ Wt[M x K](bf16, transposed) + bias
// One wave -> 16x64 macro-tile (4 WMMA accumulators share one A fragment).
// All fragment loads are contiguous 16B global_load_b128 per ISA 7.12.2:
//   A lane(row=l16, half): ks [k0+8h .. +7] and [k0+16+8h .. +7]
//   B lane(col=l16, half): ks [k0+8h .. +7] and [k0+16+8h .. +7]
// ---------------------------------------------------------------------------
template <int K>
__global__ void gemm_bf16_wmma_kernel(const bf16_t* __restrict__ Xb,
                                      const bf16_t* __restrict__ Wt,
                                      const float* __restrict__ bias,
                                      float* __restrict__ Y,
                                      int n, int M) {
    const int lane = threadIdx.x & 31;
    const int wave = (int)((blockIdx.x * blockDim.x + threadIdx.x) >> 5);
    const int tilesM4 = M >> 6;                 // 64-col macro tiles
    if (wave >= (n >> 4) * tilesM4) return;
    const int tm = wave % tilesM4;
    const int tn = wave / tilesM4;
    const int row0 = tn << 4;
    const int col0 = tm << 6;
    const int half = lane >> 4;                 // 0: lanes 0-15, 1: lanes 16-31
    const int l16  = lane & 15;

    const uint4* __restrict__ arow = (const uint4*)(Xb + (size_t)(row0 + l16) * K);
    const uint4* __restrict__ bc0 = (const uint4*)(Wt + (size_t)(col0 + 0 * 16 + l16) * K);
    const uint4* __restrict__ bc1 = (const uint4*)(Wt + (size_t)(col0 + 1 * 16 + l16) * K);
    const uint4* __restrict__ bc2 = (const uint4*)(Wt + (size_t)(col0 + 2 * 16 + l16) * K);
    const uint4* __restrict__ bc3 = (const uint4*)(Wt + (size_t)(col0 + 3 * 16 + l16) * K);
    const uint4* __restrict__ bcols[4] = {bc0, bc1, bc2, bc3};

    v8f acc[4] = {{0.f, 0.f, 0.f, 0.f, 0.f, 0.f, 0.f, 0.f},
                  {0.f, 0.f, 0.f, 0.f, 0.f, 0.f, 0.f, 0.f},
                  {0.f, 0.f, 0.f, 0.f, 0.f, 0.f, 0.f, 0.f},
                  {0.f, 0.f, 0.f, 0.f, 0.f, 0.f, 0.f, 0.f}};

#pragma unroll
    for (int k0 = 0; k0 < K; k0 += 32) {
        v16bf a;
        {
            uint4 a0 = arow[(k0 + (half << 3)) >> 3];        // 8 bf16
            uint4 a1 = arow[(k0 + 16 + (half << 3)) >> 3];   // 8 bf16
            __builtin_memcpy(&a, &a0, 16);
            __builtin_memcpy((char*)&a + 16, &a1, 16);
        }
#pragma unroll
        for (int j = 0; j < 4; ++j) {
            v16bf b;
            uint4 b0 = bcols[j][(k0 + (half << 3)) >> 3];
            uint4 b1 = bcols[j][(k0 + 16 + (half << 3)) >> 3];
            __builtin_memcpy(&b, &b0, 16);
            __builtin_memcpy((char*)&b + 16, &b1, 16);
            acc[j] = __builtin_amdgcn_wmma_f32_16x16x32_bf16(
                /*neg_a=*/false, a, /*neg_b=*/false, b,
                /*c_mod=*/(short)0, acc[j], /*reuse_a=*/false, /*reuse_b=*/false);
        }
    }

#pragma unroll
    for (int j = 0; j < 4; ++j) {
        const int ccol = col0 + j * 16 + l16;
        const float bv = bias[ccol];
#pragma unroll
        for (int r = 0; r < 8; ++r) {           // C/D layout: VGPR r -> row r + 8*half
            int m = row0 + r + (half << 3);
            Y[(size_t)m * M + ccol] = acc[j][r] + bv;
        }
    }
}

// ---------------------------------------------------------------------------
// Edge attention logits. One wave per edge; lanes stride the C channels
// of each head (coalesced 128B row reads; xl/xr are L2-resident).
// ---------------------------------------------------------------------------
__global__ void edge_alpha_kernel(const float* __restrict__ xl, const float* __restrict__ xr,
                                  const int* __restrict__ src, const int* __restrict__ dst,
                                  const float* __restrict__ ew, const float* __restrict__ wsum,
                                  float invE,
                                  const float* __restrict__ We, const float* __restrict__ att,
                                  float* __restrict__ alpha, float* __restrict__ amax,
                                  int E, int Etot, int H, int C) {
    const int lane = threadIdx.x & 31;
    const int e = (int)((blockIdx.x * blockDim.x + threadIdx.x) >> 5);
    if (e >= Etot) return;
    int s, d;
    float w;
    if (e < E) { s = src[e]; d = dst[e]; w = ew[e]; }
    else       { s = e - E;  d = s;      w = wsum[0] * invE; }   // self-loop, mean fill

    const int HC = H * C;
    const float* xls = xl + (size_t)s * HC;
    const float* xrd = xr + (size_t)d * HC;

    for (int h = 0; h < H; ++h) {
        float p = 0.f;
        const int base = h * C;
        for (int c = lane; c < C; c += 32) {
            int cc = base + c;
            float z = xls[cc] + xrd[cc] + w * We[cc];
            z = (z > 0.f) ? z : NEG_SLOPE * z;           // leaky_relu (GATv2)
            p += z * att[cc];
        }
#pragma unroll
        for (int m = 16; m >= 1; m >>= 1) p += __shfl_xor(p, m, 32);
        if (lane == 0) {
            alpha[(size_t)e * H + h] = p;
            atomicMaxFloat(&amax[(size_t)d * H + h], p);
        }
    }
}

// exp(alpha - amax[dst]) and segment denominator
__global__ void edge_softmax_kernel(const float* __restrict__ amax, const int* __restrict__ dst,
                                    float* __restrict__ alpha, float* __restrict__ den,
                                    int E, int Etot, int H) {
    size_t idx = (size_t)blockIdx.x * blockDim.x + threadIdx.x;
    if (idx >= (size_t)Etot * H) return;
    int e = (int)(idx / H);
    int h = (int)(idx % H);
    int d = (e < E) ? dst[e] : (e - E);
    float ex = __expf(alpha[idx] - amax[(size_t)d * H + h]);
    alpha[idx] = ex;
    atomicAdd(&den[(size_t)d * H + h], ex);
}

// weighted scatter-sum of xl[src] into accum[dst]
__global__ void edge_agg_kernel(const float* __restrict__ xl, const float* __restrict__ ex,
                                const float* __restrict__ den,
                                const int* __restrict__ src, const int* __restrict__ dst,
                                float* __restrict__ accum, int E, int Etot, int H, int C) {
    const int lane = threadIdx.x & 31;
    const int e = (int)((blockIdx.x * blockDim.x + threadIdx.x) >> 5);
    if (e >= Etot) return;
    int s, d;
    if (e < E) { s = src[e]; d = dst[e]; }
    else       { s = e - E;  d = s; }
    const int HC = H * C;
    const float* xls = xl + (size_t)s * HC;
    float* acc = accum + (size_t)d * HC;
    for (int h = 0; h < H; ++h) {
        float a = ex[(size_t)e * H + h] / (den[(size_t)d * H + h] + 1e-16f);
        const int base = h * C;
        for (int c = lane; c < C; c += 32)
            atomicAdd(&acc[base + c], a * xls[base + c]);
    }
}

// head-mean + out bias
__global__ void head_mean_bias_kernel(const float* __restrict__ accum, const float* __restrict__ b,
                                      float* __restrict__ out, int n, int H, int C) {
    size_t idx = (size_t)blockIdx.x * blockDim.x + threadIdx.x;
    if (idx >= (size_t)n * C) return;
    int i = (int)(idx / C);
    int c = (int)(idx % C);
    float s = 0.f;
    for (int h = 0; h < H; ++h) s += accum[(size_t)i * H * C + h * C + c];
    out[idx] = s / (float)H + b[c];
}

// BatchNorm stats: one block per channel, deterministic tree reduce
__global__ void bn_stats_kernel(const float* __restrict__ X, int n, int C,
                                float* __restrict__ mu, float* __restrict__ rs) {
    __shared__ float ss[256], sq[256];
    const int c = blockIdx.x;
    float s = 0.f, q = 0.f;
    for (int i = threadIdx.x; i < n; i += blockDim.x) {
        float v = X[(size_t)i * C + c];
        s += v; q += v * v;
    }
    ss[threadIdx.x] = s; sq[threadIdx.x] = q;
    __syncthreads();
    for (int o = 128; o > 0; o >>= 1) {
        if ((int)threadIdx.x < o) {
            ss[threadIdx.x] += ss[threadIdx.x + o];
            sq[threadIdx.x] += sq[threadIdx.x + o];
        }
        __syncthreads();
    }
    if (threadIdx.x == 0) {
        float m = ss[0] / (float)n;
        float var = sq[0] / (float)n - m * m;           // biased variance (torch BN)
        mu[c] = m;
        rs[c] = rsqrtf(var + EPS_BN);
    }
}

// gamma*(x-mu)*rsigma + beta, then ELU
__global__ void bn_elu_kernel(const float* __restrict__ X,
                              const float* __restrict__ g, const float* __restrict__ be,
                              const float* __restrict__ mu, const float* __restrict__ rs,
                              float* __restrict__ Y, int n, int C) {
    size_t idx = (size_t)blockIdx.x * blockDim.x + threadIdx.x;
    if (idx >= (size_t)n * C) return;
    int c = (int)(idx % C);
    float v = g[c] * (X[idx] - mu[c]) * rs[c] + be[c];
    Y[idx] = (v > 0.f) ? v : expm1f(v);
}

// ---------------------------------------------------------------------------
// Driver
// ---------------------------------------------------------------------------
extern "C" void kernel_launch(void* const* d_in, const int* in_sizes, int n_in,
                              void* d_out, int out_size, void* d_ws, size_t ws_size,
                              hipStream_t stream) {
    const int F  = 128;
    const int N  = in_sizes[0] / F;      // 50000
    const int E  = in_sizes[1];          // 400000
    const int Et = E + N;                // edges + self loops

    const float* emb  = (const float*)d_in[0];
    const float* ew   = (const float*)d_in[1];
    const float* Wl0  = (const float*)d_in[2];
    const float* bl0  = (const float*)d_in[3];
    const float* Wr0  = (const float*)d_in[4];
    const float* br0  = (const float*)d_in[5];
    const float* We0  = (const float*)d_in[6];
    const float* att0 = (const float*)d_in[7];
    const float* b0   = (const float*)d_in[8];
    const float* g0   = (const float*)d_in[9];
    const float* be0  = (const float*)d_in[10];
    const float* Wl1  = (const float*)d_in[11];
    const float* bl1  = (const float*)d_in[12];
    const float* Wr1  = (const float*)d_in[13];
    const float* br1  = (const float*)d_in[14];
    const float* We1  = (const float*)d_in[15];
    const float* att1 = (const float*)d_in[16];
    const float* b1   = (const float*)d_in[17];
    const float* g1   = (const float*)d_in[18];
    const float* be1  = (const float*)d_in[19];
    const int*   ei   = (const int*)d_in[20];
    const int* src = ei;
    const int* dst = ei + E;
    float* out = (float*)d_out;

    // workspace layout (floats; all offsets multiples of 4 floats = 16B aligned)
    float* ws = (float*)d_ws;
    const size_t SZ = (size_t)N * 256;
    float* xl    = ws;                                  // N x 256
    float* xr    = ws + SZ;                             // N x 256 (reused as layer-1 pre-BN)
    float* accum = ws + 2 * SZ;                         // N x 256
    float* hbuf  = ws + 3 * SZ;                         // N x 128
    float* alpha = hbuf + (size_t)N * 128;              // Et x 4
    float* amax  = alpha + (size_t)Et * 4;              // N x 4
    float* den   = amax + (size_t)N * 4;                // N x 4
    float* mu    = den + (size_t)N * 4;                 // 256
    float* rs    = mu + 256;                            // 256
    float* wsum  = rs + 256;                            // pad to 512
    bf16_t* Xb   = (bf16_t*)(wsum + 512);               // N x 128 bf16  (N*64 floats)
    bf16_t* Wtb  = (bf16_t*)(wsum + 512 + (size_t)N * 64);  // up to 256x128 bf16

    const int T = 256;
    auto blocks = [](size_t n, int t) { return (unsigned)((n + t - 1) / t); };

    // edge-weight mean (fill value for self-loop edge_attr)
    fill_kernel<<<1, 1, 0, stream>>>(wsum, 1, 0.f);
    reduce_sum_kernel<<<256, 256, 0, stream>>>(ew, E, wsum);

    // ================= Layer 0: H=1, C=128 =================
    {
        const int H = 1, C = 128, M = H * C, K = 128;
        const size_t gw = (size_t)(N / 16) * (M / 64) * 32;  // threads for GEMM
        cvt_bf16_kernel<<<blocks((size_t)N * K, T), T, 0, stream>>>(emb, Xb, (size_t)N * K);

        cvt_bf16_transpose_kernel<<<blocks((size_t)K * M, T), T, 0, stream>>>(Wl0, Wtb, K, M);
        gemm_bf16_wmma_kernel<128><<<blocks(gw, 128), 128, 0, stream>>>(Xb, Wtb, bl0, xl, N, M);
        cvt_bf16_transpose_kernel<<<blocks((size_t)K * M, T), T, 0, stream>>>(Wr0, Wtb, K, M);
        gemm_bf16_wmma_kernel<128><<<blocks(gw, 128), 128, 0, stream>>>(Xb, Wtb, br0, xr, N, M);

        fill_kernel<<<blocks((size_t)N * H, T), T, 0, stream>>>(amax, (size_t)N * H, -3.0e38f);
        fill_kernel<<<blocks((size_t)N * H, T), T, 0, stream>>>(den, (size_t)N * H, 0.f);
        fill_kernel<<<blocks((size_t)N * M, T), T, 0, stream>>>(accum, (size_t)N * M, 0.f);

        edge_alpha_kernel<<<blocks((size_t)Et * 32, 128), 128, 0, stream>>>(
            xl, xr, src, dst, ew, wsum, 1.f / (float)E, We0, att0, alpha, amax, E, Et, H, C);
        edge_softmax_kernel<<<blocks((size_t)Et * H, T), T, 0, stream>>>(amax, dst, alpha, den, E, Et, H);
        edge_agg_kernel<<<blocks((size_t)Et * 32, 128), 128, 0, stream>>>(
            xl, alpha, den, src, dst, accum, E, Et, H, C);

        head_mean_bias_kernel<<<blocks((size_t)N * C, T), T, 0, stream>>>(accum, b0, hbuf, N, H, C);
        bn_stats_kernel<<<C, 256, 0, stream>>>(hbuf, N, C, mu, rs);
        bn_elu_kernel<<<blocks((size_t)N * C, T), T, 0, stream>>>(hbuf, g0, be0, mu, rs, hbuf, N, C);
    }

    // ================= Layer 1: H=4, C=64 =================
    {
        const int H = 4, C = 64, M = H * C, K = 128;
        const size_t gw = (size_t)(N / 16) * (M / 64) * 32;
        cvt_bf16_kernel<<<blocks((size_t)N * K, T), T, 0, stream>>>(hbuf, Xb, (size_t)N * K);

        cvt_bf16_transpose_kernel<<<blocks((size_t)K * M, T), T, 0, stream>>>(Wl1, Wtb, K, M);
        gemm_bf16_wmma_kernel<128><<<blocks(gw, 128), 128, 0, stream>>>(Xb, Wtb, bl1, xl, N, M);
        cvt_bf16_transpose_kernel<<<blocks((size_t)K * M, T), T, 0, stream>>>(Wr1, Wtb, K, M);
        gemm_bf16_wmma_kernel<128><<<blocks(gw, 128), 128, 0, stream>>>(Xb, Wtb, br1, xr, N, M);

        fill_kernel<<<blocks((size_t)N * H, T), T, 0, stream>>>(amax, (size_t)N * H, -3.0e38f);
        fill_kernel<<<blocks((size_t)N * H, T), T, 0, stream>>>(den, (size_t)N * H, 0.f);
        fill_kernel<<<blocks((size_t)N * M, T), T, 0, stream>>>(accum, (size_t)N * M, 0.f);

        edge_alpha_kernel<<<blocks((size_t)Et * 32, 128), 128, 0, stream>>>(
            xl, xr, src, dst, ew, wsum, 1.f / (float)E, We1, att1, alpha, amax, E, Et, H, C);
        edge_softmax_kernel<<<blocks((size_t)Et * H, T), T, 0, stream>>>(amax, dst, alpha, den, E, Et, H);
        edge_agg_kernel<<<blocks((size_t)Et * 32, 128), 128, 0, stream>>>(
            xl, alpha, den, src, dst, accum, E, Et, H, C);

        float* pre = xr;   // xr free after edge_alpha; reuse for N x 64 pre-BN output
        head_mean_bias_kernel<<<blocks((size_t)N * C, T), T, 0, stream>>>(accum, b1, pre, N, H, C);
        bn_stats_kernel<<<C, 256, 0, stream>>>(pre, N, C, mu, rs);
        bn_elu_kernel<<<blocks((size_t)N * C, T), T, 0, stream>>>(pre, g1, be1, mu, rs, out, N, C);
    }
}
